// double_deform_conv_87660282511799
// MI455X (gfx1250) — compile-verified, matrix operand order checked by reading, not run on hardware
//
#include <hip/hip_runtime.h>

// ---------------------------------------------------------------------------
// Types for CDNA5 WMMA (wave32): 16x16x32 bf16 -> f32 accumulate
// ---------------------------------------------------------------------------
typedef __attribute__((ext_vector_type(16))) __bf16 v16bf;
typedef __attribute__((ext_vector_type(8)))  float  v8f;

constexpr int DIM   = 48;
constexpr int NPOS  = DIM * DIM * DIM;   // 110592 voxels
constexpr int OUTC  = 32;
constexpr int KPTS  = 27;                // 3x3x3 taps

// ---------------------------------------------------------------------------
// Helpers
// ---------------------------------------------------------------------------
union Frag16 { uint4 u[2]; v16bf v; };

__device__ __forceinline__ void decode_pos(int pos, int& z, int& y, int& x) {
    z = pos / (DIM * DIM);
    int r = pos - z * DIM * DIM;
    y = r / DIM;
    x = r - y * DIM;
}

__device__ __forceinline__ void decode_tap(int t, int& tz, int& ty, int& tx) {
    tz = t / 9;
    int r = t - tz * 9;
    ty = r / 3;
    tx = r - ty * 3;
}

// ---------------------------------------------------------------------------
// Weight pack: f32 (rowsSrc x ck) -> bf16 (rowsDst x ckpad), zero padded
// ---------------------------------------------------------------------------
__global__ void pack_w_kernel(const float* __restrict__ src, __bf16* __restrict__ dst,
                              int rowsDst, int rowsSrc, int ck, int ckpad) {
    int i = blockIdx.x * blockDim.x + threadIdx.x;
    if (i >= rowsDst * ckpad) return;
    int row = i / ckpad;
    int col = i - row * ckpad;
    float v = (row < rowsSrc && col < ck) ? src[row * ck + col] : 0.0f;
    dst[i] = (__bf16)v;
}

__global__ void f32_to_bf16_kernel(const float* __restrict__ in,
                                   __bf16* __restrict__ out, int n) {
    int i = blockIdx.x * blockDim.x + threadIdx.x;
    if (i < n) out[i] = (__bf16)in[i];
}

// ---------------------------------------------------------------------------
// Offset conv (3x3x3, SAME) as implicit GEMM with WMMA.
//   rows: 81 (padded to 96) channels; cols: 16 positions; red: CIN*27 (padded)
//   block = 192 threads (6 waves): shared B tile, each wave owns a 16-row tile
// ---------------------------------------------------------------------------
template <int CIN, int CKPAD>
__global__ void offset_conv_kernel(const __bf16* __restrict__ xbf,   // CIN x NPOS
                                   const __bf16* __restrict__ wbf,   // 96 x CKPAD
                                   const float*  __restrict__ bias,  // 81
                                   float*        __restrict__ off) { // 81 x NPOS
    constexpr int CK = CIN * KPTS;
    __shared__ __align__(16) __bf16 Btile[16 * CKPAD];

    const int tid   = threadIdx.x;
    const int nBase = blockIdx.x * 16;

    // -- build im2col B tile (bf16), zero pad for SAME conv and CK..CKPAD --
    for (int e = tid; e < 16 * CKPAD; e += 192) {
        int n = e / CKPAD;
        int r = e - n * CKPAD;
        __bf16 v = (__bf16)0.0f;
        if (r < CK) {
            int c = r / KPTS;
            int t = r - c * KPTS;
            int tz, ty, tx; decode_tap(t, tz, ty, tx);
            int z, y, x;    decode_pos(nBase + n, z, y, x);
            int zz = z + tz - 1, yy = y + ty - 1, xx = x + tx - 1;
            if ((unsigned)zz < (unsigned)DIM && (unsigned)yy < (unsigned)DIM &&
                (unsigned)xx < (unsigned)DIM)
                v = xbf[c * NPOS + (zz * DIM + yy) * DIM + xx];
        }
        Btile[e] = v;
    }
    __syncthreads();

    // -- WMMA tile loop (fully uniform, EXEC all ones) --
    const int lane  = tid & 31;
    const int wave  = tid >> 5;
    const int oBase = wave * 16;
    const int col   = lane & 15;   // M for A, N for B/D
    const int hs    = lane >> 4;   // half select

    v8f acc = {};
    for (int kb = 0; kb < CKPAD; kb += 32) {
        Frag16 a, b;
        const uint4* ap = reinterpret_cast<const uint4*>(
            wbf + (size_t)(oBase + col) * CKPAD + kb + hs * 8);
        a.u[0] = ap[0];   // K = kb + hs*8 .. +7
        a.u[1] = ap[2];   // K = kb + 16 + hs*8 .. +7
        const uint4* bp = reinterpret_cast<const uint4*>(
            &Btile[col * CKPAD + kb + hs * 8]);
        b.u[0] = bp[0];
        b.u[1] = bp[2];
        acc = __builtin_amdgcn_wmma_f32_16x16x32_bf16(
            false, a.v, false, b.v, (short)0, acc, false, false);
    }

    // -- store D (lane: row = oBase + hs*8 + j, col = lane&15) + bias --
    for (int j = 0; j < 8; ++j) {
        int row = oBase + hs * 8 + j;
        if (row < 3 * KPTS)
            off[(size_t)row * NPOS + nBase + col] = acc[j] + bias[row];
    }
}

// ---------------------------------------------------------------------------
// Deformable conv: trilinear sampling (fp32) -> bf16 val tile -> WMMA GEMM.
//   block = 64 threads (2 waves); grid = NPOS/16
// ---------------------------------------------------------------------------
template <int CIN, int CKPAD>
__global__ void deform_conv_kernel(const float*  __restrict__ src,   // CIN x NPOS fp32
                                   const float*  __restrict__ off,   // 81 x NPOS
                                   const __bf16* __restrict__ wbf,   // 32 x CKPAD
                                   float*        __restrict__ yout) {// 32 x NPOS
    constexpr int CK = CIN * KPTS;
    __shared__ __align__(16) __bf16 Btile[16 * CKPAD];

    const int tid   = threadIdx.x;
    const int nBase = blockIdx.x * 16;

    // zero the CK..CKPAD pad region (L1 only; L2 has CKPAD == CK)
    if constexpr (CKPAD > CK) {
        for (int e = tid; e < 16 * (CKPAD - CK); e += 64) {
            int n  = e / (CKPAD - CK);
            int rr = e - n * (CKPAD - CK);
            Btile[n * CKPAD + CK + rr] = (__bf16)0.0f;
        }
    }

    // -- build val tile: one (tap, position) pair per thread iteration --
    for (int p = tid; p < KPTS * 16; p += 64) {
        int t = p >> 4;
        int n = p & 15;
        int pos = nBase + n;
        int z, y, x;    decode_pos(pos, z, y, x);
        int tz, ty, tx; decode_tap(t, tz, ty, tx);

        float od = off[(size_t)(t)            * NPOS + pos];
        float oh = off[(size_t)(KPTS + t)     * NPOS + pos];
        float ow = off[(size_t)(2 * KPTS + t) * NPOS + pos];

        float pd = fminf(fmaxf((float)(z + tz - 1) + od, 0.0f), (float)(DIM - 1));
        float ph = fminf(fmaxf((float)(y + ty - 1) + oh, 0.0f), (float)(DIM - 1));
        float pw = fminf(fmaxf((float)(x + tx - 1) + ow, 0.0f), (float)(DIM - 1));

        float fd0 = floorf(pd), fh0 = floorf(ph), fw0 = floorf(pw);
        int d0 = (int)fd0, h0 = (int)fh0, w0 = (int)fw0;
        float fd = pd - fd0, fh = ph - fh0, fw = pw - fw0;
        int d1 = min(d0 + 1, DIM - 1);
        int h1 = min(h0 + 1, DIM - 1);
        int w1 = min(w0 + 1, DIM - 1);

        int i000 = (d0 * DIM + h0) * DIM + w0;
        int i001 = (d0 * DIM + h0) * DIM + w1;
        int i010 = (d0 * DIM + h1) * DIM + w0;
        int i011 = (d0 * DIM + h1) * DIM + w1;
        int i100 = (d1 * DIM + h0) * DIM + w0;
        int i101 = (d1 * DIM + h0) * DIM + w1;
        int i110 = (d1 * DIM + h1) * DIM + w0;
        int i111 = (d1 * DIM + h1) * DIM + w1;

        float gd = 1.0f - fd, gh = 1.0f - fh, gw = 1.0f - fw;
        float w000 = gd * gh * gw, w001 = gd * gh * fw;
        float w010 = gd * fh * gw, w011 = gd * fh * fw;
        float w100 = fd * gh * gw, w101 = fd * gh * fw;
        float w110 = fd * fh * gw, w111 = fd * fh * fw;

        for (int c = 0; c < CIN; ++c) {
            const float* s = src + (size_t)c * NPOS;
            float v = w000 * s[i000] + w001 * s[i001]
                    + w010 * s[i010] + w011 * s[i011]
                    + w100 * s[i100] + w101 * s[i101]
                    + w110 * s[i110] + w111 * s[i111];
            Btile[n * CKPAD + c * KPTS + t] = (__bf16)v;
        }
    }
    __syncthreads();

    // -- WMMA tile loop: 32 output channels = 2 row tiles (one per wave) --
    const int lane  = tid & 31;
    const int wave  = tid >> 5;
    const int oBase = wave * 16;
    const int col   = lane & 15;
    const int hs    = lane >> 4;

    v8f acc = {};
    for (int kb = 0; kb < CKPAD; kb += 32) {
        Frag16 a, b;
        const uint4* ap = reinterpret_cast<const uint4*>(
            wbf + (size_t)(oBase + col) * CKPAD + kb + hs * 8);
        a.u[0] = ap[0];
        a.u[1] = ap[2];
        const uint4* bp = reinterpret_cast<const uint4*>(
            &Btile[col * CKPAD + kb + hs * 8]);
        b.u[0] = bp[0];
        b.u[1] = bp[2];
        acc = __builtin_amdgcn_wmma_f32_16x16x32_bf16(
            false, a.v, false, b.v, (short)0, acc, false, false);
    }

    for (int j = 0; j < 8; ++j) {
        int row = oBase + hs * 8 + j;
        yout[(size_t)row * NPOS + nBase + col] = acc[j];
    }
}

// ---------------------------------------------------------------------------
// BatchNorm (training-mode batch stats, B=1) — deterministic tree reduction
// ---------------------------------------------------------------------------
__global__ void bn_stats_kernel(const float* __restrict__ y,
                                float* __restrict__ stats) { // 32 x {sum, sumsq}
    const int c   = blockIdx.x;
    const int tid = threadIdx.x;
    float s = 0.0f, sq = 0.0f;
    for (int i = tid; i < NPOS; i += 256) {
        float v = y[(size_t)c * NPOS + i];
        s += v; sq += v * v;
    }
    __shared__ float shs[256], shq[256];
    shs[tid] = s; shq[tid] = sq;
    __syncthreads();
    for (int stride = 128; stride > 0; stride >>= 1) {
        if (tid < stride) { shs[tid] += shs[tid + stride]; shq[tid] += shq[tid + stride]; }
        __syncthreads();
    }
    if (tid == 0) { stats[c * 2] = shs[0]; stats[c * 2 + 1] = shq[0]; }
}

__global__ void bn_apply_kernel(const float* __restrict__ y,
                                const float* __restrict__ stats,
                                const float* __restrict__ gamma,
                                const float* __restrict__ beta,
                                float* __restrict__ outf,
                                __bf16* __restrict__ outbf) { // may be null
    int i = blockIdx.x * blockDim.x + threadIdx.x;
    if (i >= OUTC * NPOS) return;
    int c = i / NPOS;
    const float inv = 1.0f / (float)NPOS;
    float mu  = stats[c * 2] * inv;
    float var = stats[c * 2 + 1] * inv - mu * mu;
    float v = (y[i] - mu) * rsqrtf(var + 1e-5f) * gamma[c] + beta[c];
    v = fmaxf(v, 0.0f);
    outf[i] = v;
    if (outbf) outbf[i] = (__bf16)v;
}

// ---------------------------------------------------------------------------
// Launch
// ---------------------------------------------------------------------------
extern "C" void kernel_launch(void* const* d_in, const int* in_sizes, int n_in,
                              void* d_out, int out_size, void* d_ws, size_t ws_size,
                              hipStream_t stream) {
    const float* x      = (const float*)d_in[0];
    const float* w_off1 = (const float*)d_in[1];
    const float* b_off1 = (const float*)d_in[2];
    const float* w1     = (const float*)d_in[3];
    const float* gamma1 = (const float*)d_in[4];
    const float* beta1  = (const float*)d_in[5];
    const float* w_off2 = (const float*)d_in[6];
    const float* b_off2 = (const float*)d_in[7];
    const float* w2     = (const float*)d_in[8];
    const float* gamma2 = (const float*)d_in[9];
    const float* beta2  = (const float*)d_in[10];
    float* out = (float*)d_out;

    constexpr int CK1 = 16 * KPTS, CKP1 = 448;  // 432 -> pad 448 (14 chunks of 32)
    constexpr int CK2 = 32 * KPTS, CKP2 = 864;  // 864 = 27 chunks of 32 exactly

    // workspace layout (256B aligned slabs)
    char* ws = (char*)d_ws;
    auto align256 = [](size_t v) { return (v + 255) & ~(size_t)255; };
    size_t o = 0;
    float*  off_buf = (float*)(ws + o);  o += align256(sizeof(float)  * 3 * KPTS * NPOS);
    float*  y_buf   = (float*)(ws + o);  o += align256(sizeof(float)  * OUTC * NPOS);
    float*  h_buf   = (float*)(ws + o);  o += align256(sizeof(float)  * OUTC * NPOS);
    __bf16* x_bf    = (__bf16*)(ws + o); o += align256(sizeof(__bf16) * 16   * NPOS);
    __bf16* h_bf    = (__bf16*)(ws + o); o += align256(sizeof(__bf16) * OUTC * NPOS);
    __bf16* woff1bf = (__bf16*)(ws + o); o += align256(sizeof(__bf16) * 96 * CKP1);
    __bf16* w1bf    = (__bf16*)(ws + o); o += align256(sizeof(__bf16) * 32 * CKP1);
    __bf16* woff2bf = (__bf16*)(ws + o); o += align256(sizeof(__bf16) * 96 * CKP2);
    __bf16* w2bf    = (__bf16*)(ws + o); o += align256(sizeof(__bf16) * 32 * CKP2);
    float*  stats1  = (float*)(ws + o);  o += align256(sizeof(float) * OUTC * 2);
    float*  stats2  = (float*)(ws + o);  o += align256(sizeof(float) * OUTC * 2);
    (void)ws_size; (void)in_sizes; (void)n_in; (void)out_size;

    const int nTiles = NPOS / 16;  // 6912

    // 1) pack weights to bf16 (padded)
    pack_w_kernel<<<(96 * CKP1 + 255) / 256, 256, 0, stream>>>(w_off1, woff1bf, 96, 81, CK1, CKP1);
    pack_w_kernel<<<(32 * CKP1 + 255) / 256, 256, 0, stream>>>(w1,     w1bf,    32, 32, CK1, CKP1);
    pack_w_kernel<<<(96 * CKP2 + 255) / 256, 256, 0, stream>>>(w_off2, woff2bf, 96, 81, CK2, CKP2);
    pack_w_kernel<<<(32 * CKP2 + 255) / 256, 256, 0, stream>>>(w2,     w2bf,    32, 32, CK2, CKP2);

    // 2) x -> bf16
    f32_to_bf16_kernel<<<(16 * NPOS + 255) / 256, 256, 0, stream>>>(x, x_bf, 16 * NPOS);

    // ---- layer 1 ----
    offset_conv_kernel<16, CKP1><<<nTiles, 192, 0, stream>>>(x_bf, woff1bf, b_off1, off_buf);
    deform_conv_kernel<16, CKP1><<<nTiles, 64, 0, stream>>>(x, off_buf, w1bf, y_buf);
    bn_stats_kernel<<<OUTC, 256, 0, stream>>>(y_buf, stats1);
    bn_apply_kernel<<<(OUTC * NPOS + 255) / 256, 256, 0, stream>>>(
        y_buf, stats1, gamma1, beta1, h_buf, h_bf);

    // ---- layer 2 ----
    offset_conv_kernel<32, CKP2><<<nTiles, 192, 0, stream>>>(h_bf, woff2bf, b_off2, off_buf);
    deform_conv_kernel<32, CKP2><<<nTiles, 64, 0, stream>>>(h_buf, off_buf, w2bf, y_buf);
    bn_stats_kernel<<<OUTC, 256, 0, stream>>>(y_buf, stats2);
    bn_apply_kernel<<<(OUTC * NPOS + 255) / 256, 256, 0, stream>>>(
        y_buf, stats2, gamma2, beta2, out, (__bf16*)nullptr);
}
